// VanillaEGNNLayer_83665962926202
// MI455X (gfx1250) — matrix operand order repaired and checked by reading, not compiled, not run on hardware
//
#include <hip/hip_runtime.h>

// ---------------------------------------------------------------------------
// EGNN layer for MI455X (gfx1250, wave32, WMMA f32_16x16x32_f16).
//
//  H = 128, N = 10000 nodes, E = 640000 edges.
//  All GEMMs as 16x16x32 f16 WMMA with f32 accumulation.
//  Edge kernel: each wave owns TWO 16-edge M-tiles (32 edges) so every
//  B (weight) fragment load feeds two WMMAs -> ~2x compute per vmem byte.
//  x (2.5MB f16) + weights (~230KB) are L2/WGP$-resident; gathers and the
//  82M scatter-atomics are L2 traffic, far under 23.3 TB/s HBM limits.
// ---------------------------------------------------------------------------

#define Hdim   128
#define NNODE  10000
#define NEDGE  640000

typedef __attribute__((ext_vector_type(16))) _Float16 v16h;
typedef __attribute__((ext_vector_type(8)))  _Float16 v8h;
typedef __attribute__((ext_vector_type(8)))  float    v8f;

union AFrag { v16h v; v8h p[2]; };

// 16-bit A-matrix 16x32 layout (ISA 7.12.2): lane = M (dup over halves),
// halfs t=0..7 -> K = hi*8 + t ; t=8..15 -> K = 16 + hi*8 + (t-8).
// => two contiguous 8-half (16B) loads per lane per K-chunk.
__device__ __forceinline__ v16h load_a_f16(const _Float16* base, int hi) {
  AFrag a;
  a.p[0] = *(const v8h*)(base + hi * 8);
  a.p[1] = *(const v8h*)(base + 16 + hi * 8);
  return a.v;
}

__device__ __forceinline__ v16h load_a_f32(const float* base, int hi) {
  AFrag a;
#pragma unroll
  for (int t = 0; t < 8; ++t) a.p[0][t] = (_Float16)base[hi * 8 + t];
#pragma unroll
  for (int t = 0; t < 8; ++t) a.p[1][t] = (_Float16)base[16 + hi * 8 + t];
  return a.v;
}

__device__ __forceinline__ float silu_f(float v) {
  return v / (1.0f + __expf(-v));
}

__device__ __forceinline__ v8f wmma_f16(v16h a, v16h b, v8f c) {
  return __builtin_amdgcn_wmma_f32_16x16x32_f16(false, a, false, b,
                                                (short)0, c, false, false);
}

// ---------------------------------------------------------------------------
// Utility kernels
// ---------------------------------------------------------------------------
__global__ void k_f32_to_f16(const float* __restrict__ src,
                             _Float16* __restrict__ dst, int n) {
  int i = blockIdx.x * blockDim.x + threadIdx.x;
  if (i < n) dst[i] = (_Float16)src[i];
}

__global__ void k_zero_f32(float* __restrict__ p, int n) {
  int i = blockIdx.x * blockDim.x + threadIdx.x;
  if (i < n) p[i] = 0.0f;
}

// ---------------------------------------------------------------------------
// Edge kernel: 4 waves/block, 2 M-tiles (32 edges)/wave = 128 edges/block.
// B-matrix 32x16 layout: lane&15 = N; half s -> K = (s&8)*2 + hi*8 + (s&7) (same order as A)
// -> 16 f16 loads per lane down one weight column, stride Hdim (cached).
// C/D f32 16x16: lane -> N = lane&15, vgpr r -> M = r + hi*8.
// ---------------------------------------------------------------------------
__global__ __launch_bounds__(128) void egnn_edge_kernel(
    const _Float16* __restrict__ xh, const float* __restrict__ pos,
    const int* __restrict__ erow, const int* __restrict__ ecol,
    const _Float16* __restrict__ ew1h, const float* __restrict__ ew1last,
    const float* __restrict__ eb1, const _Float16* __restrict__ ew2h,
    const float* __restrict__ eb2, const _Float16* __restrict__ cw1h,
    const float* __restrict__ cb1, const float* __restrict__ cw2,
    float* __restrict__ agg, float* __restrict__ posacc,
    float* __restrict__ deg) {
  __shared__ __align__(32) _Float16 sM[4][32 * Hdim];  // per-wave msg stage
  __shared__ float sCW[4][32];

  const int lane = threadIdx.x & 31;
  const int wv   = threadIdx.x >> 5;
  const int hi   = lane >> 4;
  const int lm   = lane & 15;

  const int ebase = blockIdx.x * 128 + wv * 32;

  // prefetch next block's edge index stream (global_prefetch_b8)
  __builtin_prefetch(&erow[ebase + 128], 0, 3);
  __builtin_prefetch(&ecol[ebase + 128], 0, 3);

  // each lane carries TWO edges: tile0 edge lm, tile1 edge 16+lm
  int   row[2], col[2];
  float px[2], py[2], pz[2], dsq[2];
#pragma unroll
  for (int t = 0; t < 2; ++t) {
    const int e = ebase + t * 16 + lm;
    row[t] = erow[e];
    col[t] = ecol[e];
    px[t] = pos[row[t] * 3 + 0] - pos[col[t] * 3 + 0];
    py[t] = pos[row[t] * 3 + 1] - pos[col[t] * 3 + 1];
    pz[t] = pos[row[t] * 3 + 2] - pos[col[t] * 3 + 2];
    dsq[t] = px[t] * px[t] + py[t] * py[t] + pz[t] * pz[t];
  }

  // broadcast per-output-row edge metadata (output M = r + hi*8)
  int   rowM[2][8];
  float dsqM[2][8];
#pragma unroll
  for (int t = 0; t < 2; ++t) {
#pragma unroll
    for (int r = 0; r < 8; ++r) {
      rowM[t][r] = __shfl(row[t], r + hi * 8);
      dsqM[t][r] = __shfl(dsq[t], r + hi * 8);
    }
  }

  // ---- layer 1: [x_i | x_j] (K=256) @ ew1[0:256] ----
  v8f acc[2][8] = {};
  const _Float16* xr0 = xh + (size_t)row[0] * Hdim;
  const _Float16* xc0 = xh + (size_t)col[0] * Hdim;
  const _Float16* xr1 = xh + (size_t)row[1] * Hdim;
  const _Float16* xc1 = xh + (size_t)col[1] * Hdim;
#pragma unroll
  for (int c = 0; c < 8; ++c) {
    const _Float16* a0p = (c < 4) ? (xr0 + c * 32) : (xc0 + (c - 4) * 32);
    const _Float16* a1p = (c < 4) ? (xr1 + c * 32) : (xc1 + (c - 4) * 32);
    v16h a0 = load_a_f16(a0p, hi);
    v16h a1 = load_a_f16(a1p, hi);
    const _Float16* brow = ew1h + (size_t)(c * 32) * Hdim + lm;
#pragma unroll
    for (int n = 0; n < 8; ++n) {
      v16h b; for (int s = 0; s < 16; ++s) b[s] = brow[(size_t)((s & 8) * 2 + hi * 8 + (s & 7)) * Hdim + n * 16];  // one B feeds two WMMAs
      acc[0][n] = wmma_f16(a0, b, acc[0][n]);
      acc[1][n] = wmma_f16(a1, b, acc[1][n]);
    }
  }
  // exact f32 rank-1 update for the dist^2 column (K=256) + bias + SiLU
#pragma unroll
  for (int n = 0; n < 8; ++n) {
    const int Nn = lm + n * 16;
    const float w256 = ew1last[Nn];
    const float b1   = eb1[Nn];
#pragma unroll
    for (int t = 0; t < 2; ++t) {
#pragma unroll
      for (int r = 0; r < 8; ++r) {
        float v = acc[t][n][r] + dsqM[t][r] * w256 + b1;
        sM[wv][(t * 16 + r + hi * 8) * Hdim + Nn] = (_Float16)silu_f(v);
      }
    }
  }
  __syncthreads();

  // ---- layer 2: m1 @ ew2 ----
  v8f acc2[2][8] = {};
  const _Float16* mrow0 = sM[wv] + lm * Hdim;
  const _Float16* mrow1 = sM[wv] + (16 + lm) * Hdim;
#pragma unroll
  for (int c = 0; c < 4; ++c) {
    v16h a0 = load_a_f16(mrow0 + c * 32, hi);
    v16h a1 = load_a_f16(mrow1 + c * 32, hi);
    const _Float16* brow = ew2h + (size_t)(c * 32) * Hdim + lm;
#pragma unroll
    for (int n = 0; n < 8; ++n) {
      v16h b; for (int s = 0; s < 16; ++s) b[s] = brow[(size_t)((s & 8) * 2 + hi * 8 + (s & 7)) * Hdim + n * 16];
      acc2[0][n] = wmma_f16(a0, b, acc2[0][n]);
      acc2[1][n] = wmma_f16(a1, b, acc2[1][n]);
    }
  }
  __syncthreads();  // all m1 reads complete before sM is overwritten

  // bias + SiLU -> message m; scatter into agg; restage for coord MLP
#pragma unroll
  for (int n = 0; n < 8; ++n) {
    const int Nn = lm + n * 16;
    const float b2 = eb2[Nn];
#pragma unroll
    for (int t = 0; t < 2; ++t) {
#pragma unroll
      for (int r = 0; r < 8; ++r) {
        float s = silu_f(acc2[t][n][r] + b2);
        sM[wv][(t * 16 + r + hi * 8) * Hdim + Nn] = (_Float16)s;
        atomicAdd(&agg[(size_t)rowM[t][r] * Hdim + Nn], s);  // no-return
      }
    }
  }
  __syncthreads();

  // ---- coord MLP: silu(m @ cw1 + cb1) . cw2 ----
  v8f acc3[2][8] = {};
#pragma unroll
  for (int c = 0; c < 4; ++c) {
    v16h a0 = load_a_f16(mrow0 + c * 32, hi);
    v16h a1 = load_a_f16(mrow1 + c * 32, hi);
    const _Float16* brow = cw1h + (size_t)(c * 32) * Hdim + lm;
#pragma unroll
    for (int n = 0; n < 8; ++n) {
      v16h b; for (int s = 0; s < 16; ++s) b[s] = brow[(size_t)((s & 8) * 2 + hi * 8 + (s & 7)) * Hdim + n * 16];
      acc3[0][n] = wmma_f16(a0, b, acc3[0][n]);
      acc3[1][n] = wmma_f16(a1, b, acc3[1][n]);
    }
  }
  float part[2][8] = {};
#pragma unroll
  for (int n = 0; n < 8; ++n) {
    const int Nn = lm + n * 16;
    const float cb = cb1[Nn];
    const float w2 = cw2[Nn];
#pragma unroll
    for (int t = 0; t < 2; ++t)
#pragma unroll
      for (int r = 0; r < 8; ++r)
        part[t][r] += silu_f(acc3[t][n][r] + cb) * w2;
  }
  // reduce over N within each 16-lane half (wave32 xor-shuffles)
#pragma unroll
  for (int t = 0; t < 2; ++t) {
#pragma unroll
    for (int r = 0; r < 8; ++r) {
#pragma unroll
      for (int m = 1; m < 16; m <<= 1)
        part[t][r] += __shfl_xor(part[t][r], m);
    }
  }
#pragma unroll
  for (int t = 0; t < 2; ++t) {
#pragma unroll
    for (int r = 0; r < 8; ++r) {
      if (lane == r)      sCW[wv][t * 16 + r]     = part[t][r];  // M=r
      if (lane == 16 + r) sCW[wv][t * 16 + 8 + r] = part[t][r];  // M=r+8
    }
  }
  __syncthreads();
  // lanes 0..15 commit tile0's edges, lanes 16..31 commit tile1's
  {
    const int t = hi;
    const float cw = sCW[wv][t * 16 + lm];
    atomicAdd(&posacc[row[t] * 3 + 0], cw * px[t]);
    atomicAdd(&posacc[row[t] * 3 + 1], cw * py[t]);
    atomicAdd(&posacc[row[t] * 3 + 2], cw * pz[t]);
    atomicAdd(&deg[row[t]], 1.0f);
  }
}

// ---------------------------------------------------------------------------
// Node kernel: h = silu([x|agg] @ nw1 + nb1) @ nw2 + nb2 ; LN(x + h).
// ---------------------------------------------------------------------------
__global__ __launch_bounds__(128) void egnn_node_kernel(
    const float* __restrict__ x, const _Float16* __restrict__ xh,
    const float* __restrict__ agg, const _Float16* __restrict__ nw1h,
    const float* __restrict__ nb1, const _Float16* __restrict__ nw2h,
    const float* __restrict__ nb2, const float* __restrict__ ln_g,
    const float* __restrict__ ln_b, float* __restrict__ outx) {
  __shared__ __align__(32) _Float16 sM[4][16 * Hdim];

  const int lane = threadIdx.x & 31;
  const int wv   = threadIdx.x >> 5;
  const int hi   = lane >> 4;
  const int lm   = lane & 15;

  const int base = blockIdx.x * 64 + wv * 16;
  const int node = min(base + lm, NNODE - 1);  // clamp tail reads

  // ---- layer 1: [x | agg] (K=256) @ nw1 ----
  v8f acc[8] = {};
  const _Float16* xr = xh + (size_t)node * Hdim;
  const float*    ar = agg + (size_t)node * Hdim;
#pragma unroll
  for (int c = 0; c < 8; ++c) {
    v16h a = (c < 4) ? load_a_f16(xr + c * 32, hi)
                     : load_a_f32(ar + (c - 4) * 32, hi);
    const _Float16* brow = nw1h + (size_t)(c * 32) * Hdim + lm;
#pragma unroll
    for (int n = 0; n < 8; ++n) {
      v16h b; for (int s = 0; s < 16; ++s) b[s] = brow[(size_t)((s & 8) * 2 + hi * 8 + (s & 7)) * Hdim + n * 16];
      acc[n] = wmma_f16(a, b, acc[n]);
    }
  }
#pragma unroll
  for (int n = 0; n < 8; ++n) {
    const int Nn = lm + n * 16;
    const float b1 = nb1[Nn];
#pragma unroll
    for (int r = 0; r < 8; ++r)
      sM[wv][(r + hi * 8) * Hdim + Nn] = (_Float16)silu_f(acc[n][r] + b1);
  }
  __syncthreads();

  // ---- layer 2 ----
  v8f acc2[8] = {};
  const _Float16* mrow = sM[wv] + lm * Hdim;
#pragma unroll
  for (int c = 0; c < 4; ++c) {
    v16h a = load_a_f16(mrow + c * 32, hi);
    const _Float16* brow = nw2h + (size_t)(c * 32) * Hdim + lm;
#pragma unroll
    for (int n = 0; n < 8; ++n) {
      v16h b; for (int s = 0; s < 16; ++s) b[s] = brow[(size_t)((s & 8) * 2 + hi * 8 + (s & 7)) * Hdim + n * 16];
      acc2[n] = wmma_f16(a, b, acc2[n]);
    }
  }

  // ---- residual + LayerNorm (f32) ----
  float s1[8] = {}, s2[8] = {};
#pragma unroll
  for (int n = 0; n < 8; ++n) {
    const int Nn = lm + n * 16;
    const float b2 = nb2[Nn];
#pragma unroll
    for (int r = 0; r < 8; ++r) {
      const int nm = min(base + r + hi * 8, NNODE - 1);
      float y = x[(size_t)nm * Hdim + Nn] + acc2[n][r] + b2;
      acc2[n][r] = y;
      s1[r] += y;
      s2[r] += y * y;
    }
  }
#pragma unroll
  for (int r = 0; r < 8; ++r) {
#pragma unroll
    for (int m = 1; m < 16; m <<= 1) {
      s1[r] += __shfl_xor(s1[r], m);
      s2[r] += __shfl_xor(s2[r], m);
    }
  }
#pragma unroll
  for (int n = 0; n < 8; ++n) {
    const int Nn = lm + n * 16;
    const float g  = ln_g[Nn];
    const float bb = ln_b[Nn];
#pragma unroll
    for (int r = 0; r < 8; ++r) {
      const int nm = base + r + hi * 8;
      if (nm < NNODE) {
        const float mu  = s1[r] * (1.0f / Hdim);
        const float var = s2[r] * (1.0f / Hdim) - mu * mu;
        outx[(size_t)nm * Hdim + Nn] =
            (acc2[n][r] - mu) * rsqrtf(var + 1e-5f) * g + bb;
      }
    }
  }
}

__global__ void egnn_pos_kernel(const float* __restrict__ pos,
                                const float* __restrict__ posacc,
                                const float* __restrict__ deg,
                                float* __restrict__ outp) {
  int i = blockIdx.x * blockDim.x + threadIdx.x;
  if (i < NNODE) {
    const float d = deg[i] + 1e-6f;
    outp[i * 3 + 0] = pos[i * 3 + 0] + posacc[i * 3 + 0] / d;
    outp[i * 3 + 1] = pos[i * 3 + 1] + posacc[i * 3 + 1] / d;
    outp[i * 3 + 2] = pos[i * 3 + 2] + posacc[i * 3 + 2] / d;
  }
}

// ---------------------------------------------------------------------------
extern "C" void kernel_launch(void* const* d_in, const int* in_sizes, int n_in,
                              void* d_out, int out_size, void* d_ws,
                              size_t ws_size, hipStream_t stream) {
  const float* x    = (const float*)d_in[0];
  const float* pos  = (const float*)d_in[1];
  const int*   ei   = (const int*)d_in[2];  // [2, E]
  const float* ew1  = (const float*)d_in[3];
  const float* eb1  = (const float*)d_in[4];
  const float* ew2  = (const float*)d_in[5];
  const float* eb2  = (const float*)d_in[6];
  const float* nw1  = (const float*)d_in[7];
  const float* nb1  = (const float*)d_in[8];
  const float* nw2  = (const float*)d_in[9];
  const float* nb2  = (const float*)d_in[10];
  const float* cw1  = (const float*)d_in[11];
  const float* cb1  = (const float*)d_in[12];
  const float* cw2  = (const float*)d_in[13];
  const float* ln_g = (const float*)d_in[14];
  const float* ln_b = (const float*)d_in[15];

  // workspace carve (all offsets 16B aligned)
  char* w = (char*)d_ws;
  size_t off = 0;
  _Float16* xh   = (_Float16*)(w + off); off += (size_t)NNODE * Hdim * 2;
  _Float16* ew1h = (_Float16*)(w + off); off += (size_t)256 * Hdim * 2;
  _Float16* ew2h = (_Float16*)(w + off); off += (size_t)Hdim * Hdim * 2;
  _Float16* nw1h = (_Float16*)(w + off); off += (size_t)256 * Hdim * 2;
  _Float16* nw2h = (_Float16*)(w + off); off += (size_t)Hdim * Hdim * 2;
  _Float16* cw1h = (_Float16*)(w + off); off += (size_t)Hdim * Hdim * 2;
  float* agg    = (float*)(w + off); off += (size_t)NNODE * Hdim * 4;
  float* posacc = (float*)(w + off); off += (size_t)NNODE * 3 * 4;
  float* deg    = (float*)(w + off); off += (size_t)NNODE * 4;

  auto cdiv = [](int a, int b) { return (a + b - 1) / b; };

  // f32 -> f16 weight/feature conversion
  k_f32_to_f16<<<cdiv(NNODE * Hdim, 256), 256, 0, stream>>>(x, xh, NNODE * Hdim);
  k_f32_to_f16<<<cdiv(256 * Hdim, 256), 256, 0, stream>>>(ew1, ew1h, 256 * Hdim);
  k_f32_to_f16<<<cdiv(Hdim * Hdim, 256), 256, 0, stream>>>(ew2, ew2h, Hdim * Hdim);
  k_f32_to_f16<<<cdiv(256 * Hdim, 256), 256, 0, stream>>>(nw1, nw1h, 256 * Hdim);
  k_f32_to_f16<<<cdiv(Hdim * Hdim, 256), 256, 0, stream>>>(nw2, nw2h, Hdim * Hdim);
  k_f32_to_f16<<<cdiv(Hdim * Hdim, 256), 256, 0, stream>>>(cw1, cw1h, Hdim * Hdim);
  // agg/posacc/deg are contiguous: one zero pass (required every replay)
  k_zero_f32<<<cdiv(NNODE * (Hdim + 4), 256), 256, 0, stream>>>(
      agg, NNODE * (Hdim + 4));

  const float* ew1last = ew1 + (size_t)256 * Hdim;  // dist^2 row of ew1 (f32)

  egnn_edge_kernel<<<NEDGE / 128, 128, 0, stream>>>(
      xh, pos, ei, ei + NEDGE, ew1h, ew1last, eb1, ew2h, eb2, cw1h, cb1, cw2,
      agg, posacc, deg);

  float* outx = (float*)d_out;
  float* outp = outx + (size_t)NNODE * Hdim;

  egnn_node_kernel<<<cdiv(NNODE, 64), 128, 0, stream>>>(
      x, xh, agg, nw1h, nb1, nw2h, nb2, ln_g, ln_b, outx);
  egnn_pos_kernel<<<cdiv(NNODE, 256), 256, 0, stream>>>(pos, posacc, deg, outp);
}
